// HomographyNetWithDSAC_76287209111876
// MI455X (gfx1250) — compile-verified, matrix operand order checked by reading, not run on hardware
//
#include <hip/hip_runtime.h>
#include <hip/hip_bf16.h>

typedef __attribute__((ext_vector_type(16))) _Float16 v16h;
typedef __attribute__((ext_vector_type(8)))  _Float16 v8h;
typedef __attribute__((ext_vector_type(8)))  float    v8f;

#define BATCH 16
#define IMGP  128
#define BN_INV 0.9999950000374997f   // 1/sqrt(1+1e-5)
#define LSTR 40   // LDS row stride (halfs); 80B rows keep 16B alignment, skew banks

#if __has_builtin(__builtin_amdgcn_global_load_async_to_lds_b128) && \
    __has_builtin(__builtin_amdgcn_s_wait_asynccnt)
#define USE_ASYNC_LDS 1
typedef int vi4 __attribute__((vector_size(16)));
typedef __attribute__((address_space(1))) vi4 gvi4;   // global int4
typedef __attribute__((address_space(3))) vi4 lvi4;   // LDS int4
#else
#define USE_ASYNC_LDS 0
#endif

// ---------------------------------------------------------------------------
// Zero a small pad region (OOB staging reads are redirected here)
// ---------------------------------------------------------------------------
__global__ void zero_region(float* __restrict__ p) {
    p[threadIdx.x] = 0.0f;
}

// ---------------------------------------------------------------------------
// Input NCHW f32 -> NHWC f16   (x: (16,2,128,128) -> (16,128,128,2))
// ---------------------------------------------------------------------------
__global__ void cvt_x_nhwc(const float* __restrict__ in, _Float16* __restrict__ out, int n) {
    int i = blockIdx.x * blockDim.x + threadIdx.x;
    if (i >= n) return;
    int c = i & 1;  int pos = i >> 1;
    int x = pos & 127; pos >>= 7;
    int y = pos & 127; int b = pos >> 7;
    out[i] = (_Float16)in[(((size_t)b * 2 + c) * IMGP + y) * IMGP + x];
}

// ---------------------------------------------------------------------------
// Conv weights f32 [O][Cin][3][3] -> f16 [O][r][Cin]  (r = ky*3+kx)
// ---------------------------------------------------------------------------
__global__ void repack_w(const float* __restrict__ in, _Float16* __restrict__ out,
                         int Cin, int n) {
    int i = blockIdx.x * blockDim.x + threadIdx.x;
    if (i >= n) return;
    int K = Cin * 9;
    int o = i / K; int rem = i - o * K;
    int r = rem / Cin; int ci = rem - r * Cin;
    out[i] = (_Float16)in[((size_t)o * Cin + ci) * 9 + r];
}

// ---------------------------------------------------------------------------
// Fast implicit-GEMM conv (NHWC, Cin % 32 == 0), async double-buffered staging
//   M = Cout tile 64, N = pixel tile 64, K = 9*Cin in steps of 32 (one tap r)
//   K-step s: weight addr = base + 32*s (contiguous); tap r = s >> cishift
// ---------------------------------------------------------------------------
__global__ __launch_bounds__(128)
void conv3x3_wmma_nhwc(const _Float16* __restrict__ in, _Float16* __restrict__ out,
                       const _Float16* __restrict__ w16,
                       const _Float16* __restrict__ zpad,
                       const float* __restrict__ cb, const float* __restrict__ bg,
                       const float* __restrict__ bb,
                       int Cin, int Cout, int H, int W, int cishift)
{
    __shared__ _Float16 Wt[2][64][LSTR];
    __shared__ _Float16 Pt[2][64][LSTR];

    const int tid  = threadIdx.x;
    const int wave = tid >> 5;
    const int lane = tid & 31;
    const int HW   = H * W;
    const int ptile0 = blockIdx.x * 64;
    const int otile0 = blockIdx.y * 64;
    const int Ktot   = Cin * 9;          // multiple of 32
    const int nsteps = Ktot >> 5;

    // staging coordinates (thread -> one 32B chunk of one LDS row)
    const int srow = tid >> 1;
    const int skb  = (tid & 1) * 16;
    const int sgp  = ptile0 + srow;
    const int sb   = sgp / HW;
    const int srem = sgp - sb * HW;
    const int sy   = srem / W;
    const int sx   = srem - sy * W;
    const _Float16* wrow = w16 + (size_t)(otile0 + srow) * Ktot + skb;

    // stage K-step s into LDS buffer bufi (branch-free; OOB -> zero pad)
    auto stage = [&](int s, int bufi) {
        const int r   = s >> cishift;
        const int ci0 = (s - (r << cishift)) << 5;
        const int ry  = r / 3;
        const int rx  = r - ry * 3;
        const int yy  = sy + ry - 1;
        const int xx  = sx + rx - 1;
        const bool inb = (yy >= 0) & (yy < H) & (xx >= 0) & (xx < W);
        const _Float16* wsrc = wrow + (s << 5);
        const _Float16* psrc = inb
            ? (in + ((size_t)(sb * H + yy) * W + xx) * Cin + ci0 + skb)
            : (zpad + skb);
#if USE_ASYNC_LDS
        __builtin_amdgcn_global_load_async_to_lds_b128(
            (gvi4*)wsrc,       (lvi4*)&Wt[bufi][srow][skb],     0, 0);
        __builtin_amdgcn_global_load_async_to_lds_b128(
            (gvi4*)(wsrc + 8), (lvi4*)&Wt[bufi][srow][skb + 8], 0, 0);
        __builtin_amdgcn_global_load_async_to_lds_b128(
            (gvi4*)psrc,       (lvi4*)&Pt[bufi][srow][skb],     0, 0);
        __builtin_amdgcn_global_load_async_to_lds_b128(
            (gvi4*)(psrc + 8), (lvi4*)&Pt[bufi][srow][skb + 8], 0, 0);
#else
        const v8h* wsv = reinterpret_cast<const v8h*>(wsrc);
        const v8h* psv = reinterpret_cast<const v8h*>(psrc);
        v8h w0 = wsv[0], w1 = wsv[1];
        v8h p0 = psv[0], p1 = psv[1];
        *reinterpret_cast<v8h*>(&Wt[bufi][srow][skb])     = w0;
        *reinterpret_cast<v8h*>(&Wt[bufi][srow][skb + 8]) = w1;
        *reinterpret_cast<v8h*>(&Pt[bufi][srow][skb])     = p0;
        *reinterpret_cast<v8h*>(&Pt[bufi][srow][skb + 8]) = p1;
#endif
    };

    v8f acc[4];
    #pragma unroll
    for (int t = 0; t < 4; ++t)
        #pragma unroll
        for (int j = 0; j < 8; ++j) acc[t][j] = 0.0f;

    const int rsel = lane & 15;
    const int half = lane >> 4;

    stage(0, 0);
#if USE_ASYNC_LDS
    __builtin_amdgcn_s_wait_asynccnt(0);
#endif
    __syncthreads();

    for (int s = 0; s < nsteps; ++s) {
        const int cur = s & 1;
        if (s + 1 < nsteps) stage(s + 1, cur ^ 1);   // overlap with compute

        v8h alo = *reinterpret_cast<const v8h*>(&Wt[cur][wave * 16 + rsel][half * 8]);
        v8h ahi = *reinterpret_cast<const v8h*>(&Wt[cur][wave * 16 + rsel][16 + half * 8]);
        v16h a  = __builtin_shufflevector(alo, ahi, 0,1,2,3,4,5,6,7,8,9,10,11,12,13,14,15);
        #pragma unroll
        for (int t = 0; t < 4; ++t) {
            v8h blo = *reinterpret_cast<const v8h*>(&Pt[cur][t * 16 + rsel][half * 8]);
            v8h bhi = *reinterpret_cast<const v8h*>(&Pt[cur][t * 16 + rsel][16 + half * 8]);
            v16h bf = __builtin_shufflevector(blo, bhi, 0,1,2,3,4,5,6,7,8,9,10,11,12,13,14,15);
            acc[t] = __builtin_amdgcn_wmma_f32_16x16x32_f16(
                         false, a, false, bf, (short)0, acc[t], false, false);
        }
#if USE_ASYNC_LDS
        __builtin_amdgcn_s_wait_asynccnt(0);
#endif
        __syncthreads();
    }

    // ---- epilogue: 8 consecutive channels per lane -> one b128 store per tile
    const int mofs  = (lane < 16) ? 0 : 8;
    const int obase = otile0 + wave * 16 + mofs;
    float sc[8], c0[8], b0[8];
    #pragma unroll
    for (int j = 0; j < 8; ++j) {
        sc[j] = bg[obase + j] * BN_INV;
        c0[j] = cb[obase + j];
        b0[j] = bb[obase + j];
    }
    #pragma unroll
    for (int t = 0; t < 4; ++t) {
        const int gp  = ptile0 + t * 16 + rsel;
        const int b   = gp / HW;
        const int rem = gp - b * HW;
        const int y   = rem / W;
        const int x   = rem - y * W;
        v8h ov;
        #pragma unroll
        for (int j = 0; j < 8; ++j) {
            float v = (acc[t][j] + c0[j]) * sc[j] + b0[j];
            v = v > 0.0f ? v : 0.0f;
            ov[j] = (_Float16)v;
        }
        *reinterpret_cast<v8h*>(
            &out[((size_t)(b * H + y) * W + x) * Cout + obase]) = ov;
    }
}

// ---------------------------------------------------------------------------
// Generic conv (layer 0 only: Cin=2, Ktot=18, single K-step) — NHWC layouts
// ---------------------------------------------------------------------------
__global__ __launch_bounds__(128)
void conv3x3_wmma_generic(const _Float16* __restrict__ in, _Float16* __restrict__ out,
                          const _Float16* __restrict__ w16,
                          const float* __restrict__ cb, const float* __restrict__ bg,
                          const float* __restrict__ bb,
                          int Cin, int Cout, int H, int W)
{
    __shared__ _Float16 Wt[64][LSTR];
    __shared__ _Float16 Pt[64][LSTR];

    const int tid  = threadIdx.x;
    const int wave = tid >> 5;
    const int lane = tid & 31;
    const int HW   = H * W;
    const int ptile0 = blockIdx.x * 64;
    const int otile0 = blockIdx.y * 64;
    const int Ktot   = Cin * 9;

    v8f acc[4];
    #pragma unroll
    for (int t = 0; t < 4; ++t)
        #pragma unroll
        for (int j = 0; j < 8; ++j) acc[t][j] = 0.0f;

    const int rsel = lane & 15;
    const int half = lane >> 4;

    for (int k0 = 0; k0 < Ktot; k0 += 32) {
        {
            const int row = tid >> 1;
            const int kb  = (tid & 1) * 16;
            const int o   = otile0 + row;
            #pragma unroll
            for (int i = 0; i < 16; ++i) {
                int k = k0 + kb + i;
                _Float16 v = (_Float16)0.0f;
                if (k < Ktot) v = w16[(size_t)o * Ktot + k];
                Wt[row][kb + i] = v;
            }
        }
        {
            const int prow = tid >> 1;
            const int kb   = (tid & 1) * 16;
            const int gp   = ptile0 + prow;
            const int b    = gp / HW;
            const int rem  = gp - b * HW;
            const int y    = rem / W;
            const int x    = rem - y * W;
            #pragma unroll
            for (int i = 0; i < 16; ++i) {
                int k = k0 + kb + i;
                _Float16 v = (_Float16)0.0f;
                if (k < Ktot) {
                    int r  = k / Cin;
                    int ci = k - r * Cin;
                    int ry = r / 3;
                    int yy = y + ry - 1, xx = x + (r - ry * 3) - 1;
                    if (yy >= 0 && yy < H && xx >= 0 && xx < W)
                        v = in[((size_t)(b * H + yy) * W + xx) * Cin + ci];
                }
                Pt[prow][kb + i] = v;
            }
        }
        __syncthreads();

        v8h alo = *reinterpret_cast<const v8h*>(&Wt[wave * 16 + rsel][half * 8]);
        v8h ahi = *reinterpret_cast<const v8h*>(&Wt[wave * 16 + rsel][16 + half * 8]);
        v16h a  = __builtin_shufflevector(alo, ahi, 0,1,2,3,4,5,6,7,8,9,10,11,12,13,14,15);
        #pragma unroll
        for (int t = 0; t < 4; ++t) {
            v8h blo = *reinterpret_cast<const v8h*>(&Pt[t * 16 + rsel][half * 8]);
            v8h bhi = *reinterpret_cast<const v8h*>(&Pt[t * 16 + rsel][16 + half * 8]);
            v16h bf = __builtin_shufflevector(blo, bhi, 0,1,2,3,4,5,6,7,8,9,10,11,12,13,14,15);
            acc[t] = __builtin_amdgcn_wmma_f32_16x16x32_f16(
                         false, a, false, bf, (short)0, acc[t], false, false);
        }
        __syncthreads();
    }

    const int mofs  = (lane < 16) ? 0 : 8;
    const int obase = otile0 + wave * 16 + mofs;
    float sc[8], c0[8], b0[8];
    #pragma unroll
    for (int j = 0; j < 8; ++j) {
        sc[j] = bg[obase + j] * BN_INV;
        c0[j] = cb[obase + j];
        b0[j] = bb[obase + j];
    }
    #pragma unroll
    for (int t = 0; t < 4; ++t) {
        const int gp  = ptile0 + t * 16 + rsel;
        const int b   = gp / HW;
        const int rem = gp - b * HW;
        const int y   = rem / W;
        const int x   = rem - y * W;
        v8h ov;
        #pragma unroll
        for (int j = 0; j < 8; ++j) {
            float v = (acc[t][j] + c0[j]) * sc[j] + b0[j];
            v = v > 0.0f ? v : 0.0f;
            ov[j] = (_Float16)v;
        }
        *reinterpret_cast<v8h*>(
            &out[((size_t)(b * H + y) * W + x) * Cout + obase]) = ov;
    }
}

// ---------------------------------------------------------------------------
// 2x2 maxpool, NHWC f16 (coalesced across channels)
// ---------------------------------------------------------------------------
__global__ void maxpool2_nhwc(const _Float16* __restrict__ in, _Float16* __restrict__ out,
                              int C, int Hin)
{
    const int Hout = Hin >> 1;
    const size_t n = (size_t)BATCH * C * Hout * Hout;
    size_t i = (size_t)blockIdx.x * blockDim.x + threadIdx.x;
    if (i >= n) return;
    int c = (int)(i % C); size_t r = i / C;
    int x = (int)(r % Hout); r /= Hout;
    int y = (int)(r % Hout); int b = (int)(r / Hout);
    const _Float16* p = &in[((size_t)(b * Hin + 2 * y) * Hin + 2 * x) * C + c];
    float m = fmaxf(fmaxf((float)p[0], (float)p[C]),
                    fmaxf((float)p[(size_t)Hin * C], (float)p[(size_t)Hin * C + C]));
    out[i] = (_Float16)m;
}

// ---------------------------------------------------------------------------
// Features NHWC (16,16,16,128) -> NCHW flat f16 (matches reshape(B,-1))
// ---------------------------------------------------------------------------
__global__ void repack_feat(const _Float16* __restrict__ in, _Float16* __restrict__ out, int n) {
    int i = blockIdx.x * blockDim.x + threadIdx.x;
    if (i >= n) return;
    int x = i & 15;  int r = i >> 4;
    int y = r & 15;  r >>= 4;
    int c = r & 127; int b = r >> 7;
    out[i] = in[((size_t)((b * 16 + y) * 16 + x)) * 128 + c];
}

// ---------------------------------------------------------------------------
// FC1: h1[16x1024] = feat[16x32768](f16) @ W^T(f32, converted inline) + b
// ---------------------------------------------------------------------------
__global__ __launch_bounds__(128)
void fc1_wmma(const _Float16* __restrict__ feat, const float* __restrict__ w,
              const float* __restrict__ bias, float* __restrict__ h1)
{
    __shared__ float red[4][16][16];
    const int tid  = threadIdx.x;
    const int wave = tid >> 5;
    const int lane = tid & 31;
    const int rsel = lane & 15;
    const int half = lane >> 4;
    const int m0   = blockIdx.x * 16;

    v8f acc;
    #pragma unroll
    for (int j = 0; j < 8; ++j) acc[j] = 0.0f;

    const int kbeg = wave * 8192;
    const int kend = kbeg + 8192;
    const float*    wrow = w    + (size_t)(m0 + rsel) * 32768;
    const _Float16* frow = feat + (size_t)rsel * 32768;

    for (int k0 = kbeg; k0 < kend; k0 += 32) {
        __builtin_prefetch(wrow + k0 + 256, 0, 0);   // global_prefetch_b8
        union { _Float16 h[16]; v16h v; } ua;
        #pragma unroll
        for (int i = 0; i < 8; ++i) ua.h[i]     = (_Float16)wrow[k0 + half * 8 + i];
        #pragma unroll
        for (int i = 0; i < 8; ++i) ua.h[8 + i] = (_Float16)wrow[k0 + 16 + half * 8 + i];

        v8h blo = *reinterpret_cast<const v8h*>(&frow[k0 + half * 8]);
        v8h bhi = *reinterpret_cast<const v8h*>(&frow[k0 + 16 + half * 8]);
        v16h bf = __builtin_shufflevector(blo, bhi, 0,1,2,3,4,5,6,7,8,9,10,11,12,13,14,15);

        acc = __builtin_amdgcn_wmma_f32_16x16x32_f16(
                  false, ua.v, false, bf, (short)0, acc, false, false);
    }

    const int mofs = (lane < 16) ? 0 : 8;
    #pragma unroll
    for (int j = 0; j < 8; ++j)
        red[wave][mofs + j][rsel] = acc[j];
    __syncthreads();

    for (int e = tid; e < 256; e += 128) {
        int m = e >> 4, n = e & 15;
        float s = red[0][m][n] + red[1][m][n] + red[2][m][n] + red[3][m][n] + bias[m0 + m];
        h1[(size_t)n * 1024 + m0 + m] = s;
    }
}

// ---------------------------------------------------------------------------
// FC2 head + cnn_H assembly
// ---------------------------------------------------------------------------
__global__ void fc2_head(const float* __restrict__ h1, const float* __restrict__ w,
                         const float* __restrict__ b, float* __restrict__ cnnH)
{
    const int t = threadIdx.x;       // 128 threads = 16*8
    const int bb = t >> 3, o = t & 7;
    const float* hv = h1 + (size_t)bb * 1024;
    const float* wv = w  + (size_t)o * 1024;
    float s = b[o];
    for (int i = 0; i < 1024; ++i) s += hv[i] * wv[i];
    cnnH[bb * 9 + o] = s;
    if (o == 0) cnnH[bb * 9 + 8] = 1.0f;
}

// ---------------------------------------------------------------------------
// DLT + 9x9 Jacobi eigensolver, one thread per hypothesis
// ---------------------------------------------------------------------------
__global__ __launch_bounds__(256)
void dlt_jacobi(const float* __restrict__ src, const float* __restrict__ dst,
                float* __restrict__ hyps)
{
    const int idx = blockIdx.x * blockDim.x + threadIdx.x;
    if (idx >= BATCH * 256) return;
    const float* sp = src + (size_t)idx * 8;
    const float* dp = dst + (size_t)idx * 8;

    float M[9][9] = {};
    for (int p = 0; p < 4; ++p) {
        float x = sp[2 * p], y = sp[2 * p + 1];
        float u = dp[2 * p], v = dp[2 * p + 1];
        float r1[9] = { x, y, 1.f, 0.f, 0.f, 0.f, -u * x, -u * y, -u };
        float r2[9] = { 0.f, 0.f, 0.f, x, y, 1.f, -v * x, -v * y, -v };
        for (int i = 0; i < 9; ++i)
            for (int j = 0; j < 9; ++j)
                M[i][j] += r1[i] * r1[j] + r2[i] * r2[j];
    }

    float V[9][9];
    for (int i = 0; i < 9; ++i)
        for (int j = 0; j < 9; ++j)
            V[i][j] = (i == j) ? 1.0f : 0.0f;

    for (int sweep = 0; sweep < 12; ++sweep) {
        for (int p = 0; p < 8; ++p) {
            for (int q = p + 1; q < 9; ++q) {
                float apq = M[p][q];
                float phi = 0.5f * atan2f(2.0f * apq, M[q][q] - M[p][p]);
                float c = cosf(phi), s = sinf(phi);
                for (int i = 0; i < 9; ++i) {
                    float mip = M[i][p], miq = M[i][q];
                    M[i][p] = c * mip - s * miq;
                    M[i][q] = s * mip + c * miq;
                }
                for (int i = 0; i < 9; ++i) {
                    float mpi = M[p][i], mqi = M[q][i];
                    M[p][i] = c * mpi - s * mqi;
                    M[q][i] = s * mpi + c * mqi;
                }
                for (int i = 0; i < 9; ++i) {
                    float vip = V[i][p], viq = V[i][q];
                    V[i][p] = c * vip - s * viq;
                    V[i][q] = s * vip + c * viq;
                }
            }
        }
    }

    int mi = 0; float mv = M[0][0];
    for (int i = 1; i < 9; ++i) if (M[i][i] < mv) { mv = M[i][i]; mi = i; }
    float invd = 1.0f / (V[8][mi] + 1e-8f);
    for (int i = 0; i < 9; ++i)
        hyps[(size_t)idx * 9 + i] = V[i][mi] * invd;
}

// ---------------------------------------------------------------------------
// Hypothesis scoring: bilinear warp of p1 (LDS) vs p2; one wave per hypothesis
// ---------------------------------------------------------------------------
__device__ __forceinline__ float samp_border0(const float* img, float xi, float yi) {
    bool inb = (xi >= 0.f) & (xi <= 127.f) & (yi >= 0.f) & (yi <= 127.f);
    int xc = (int)fminf(fmaxf(xi, 0.f), 127.f);
    int yc = (int)fminf(fmaxf(yi, 0.f), 127.f);
    float v = img[yc * IMGP + xc];
    return inb ? v : 0.0f;
}

__global__ __launch_bounds__(256)
void dsac_score(const float* __restrict__ x, const float* __restrict__ hyps,
                float* __restrict__ scores)
{
    __shared__ float sP1[IMGP * IMGP];   // 64 KB of the 320 KB/WGP LDS
    const int tid  = threadIdx.x;
    const int wave = tid >> 5;
    const int lane = tid & 31;
    const int b    = blockIdx.x >> 5;
    const int grp  = blockIdx.x & 31;

    const float* p1 = x + (size_t)b * 2 * IMGP * IMGP;
    const float* p2 = p1 + IMGP * IMGP;

    for (int i = tid; i < IMGP * IMGP; i += 256) sP1[i] = p1[i];
    __syncthreads();

    const int k = grp * 8 + wave;
    const float* H = hyps + ((size_t)b * 256 + k) * 9;
    const float t00 = H[0], t01 = H[1], t02 = H[2];
    const float t10 = H[3], t11 = H[4], t12 = H[5];

    float acc = 0.0f;
    const float step = 2.0f / 127.0f;
    for (int pix = lane; pix < IMGP * IMGP; pix += 32) {
        int py = pix >> 7, px = pix & 127;
        float X = -1.0f + px * step;
        float Y = -1.0f + py * step;
        float gx = t00 * X + t01 * Y + t02;
        float gy = t10 * X + t11 * Y + t12;
        float fx = (gx + 1.0f) * 0.5f * 127.0f;
        float fy = (gy + 1.0f) * 0.5f * 127.0f;
        float x0 = floorf(fx), y0 = floorf(fy);
        float wx = fx - x0,   wy = fy - y0;
        float v00 = samp_border0(sP1, x0,       y0);
        float v10 = samp_border0(sP1, x0 + 1.f, y0);
        float v01 = samp_border0(sP1, x0,       y0 + 1.f);
        float v11 = samp_border0(sP1, x0 + 1.f, y0 + 1.f);
        float wv = v00 * (1.f - wx) * (1.f - wy) + v10 * wx * (1.f - wy)
                 + v01 * (1.f - wx) * wy         + v11 * wx * wy;
        acc += fabsf(wv - p2[pix]);
    }
    #pragma unroll
    for (int off = 16; off > 0; off >>= 1)
        acc += __shfl_down(acc, off, 32);
    if (lane == 0)
        scores[b * 256 + k] = -acc * (1.0f / 16384.0f);
}

// ---------------------------------------------------------------------------
// Argmax select + blend
// ---------------------------------------------------------------------------
__global__ void dsac_select(const float* __restrict__ scores, const float* __restrict__ hyps,
                            const float* __restrict__ cnnH, float* __restrict__ out)
{
    const int bb = threadIdx.x;
    if (bb >= BATCH) return;
    const float* s = scores + bb * 256;
    int sel = 0; float best = s[0];
    for (int kk = 1; kk < 256; ++kk)
        if (s[kk] > best) { best = s[kk]; sel = kk; }
    const float* hv = hyps + ((size_t)bb * 256 + sel) * 9;
    const float* cv = cnnH + bb * 9;
    for (int i = 0; i < 9; ++i)
        out[bb * 9 + i] = 0.5f * (cv[i] + hv[i]);
}

// ---------------------------------------------------------------------------
// Host-side orchestration
// ---------------------------------------------------------------------------
static inline size_t alignup(size_t v) { return (v + 255) & ~(size_t)255; }

extern "C" void kernel_launch(void* const* d_in, const int* in_sizes, int n_in,
                              void* d_out, int out_size, void* d_ws, size_t ws_size,
                              hipStream_t stream)
{
    (void)in_sizes; (void)n_in; (void)out_size; (void)ws_size;

    const float* x       = (const float*)d_in[0];
    const float* src_pts = (const float*)d_in[1];
    const float* dst_pts = (const float*)d_in[2];
    const float* fc1_w = (const float*)d_in[35];
    const float* fc1_b = (const float*)d_in[36];
    const float* fc2_w = (const float*)d_in[37];
    const float* fc2_b = (const float*)d_in[38];
    float* out = (float*)d_out;

    // ---- workspace layout
    char* ws = (char*)d_ws;
    size_t off = 0;
    _Float16* x16  = (_Float16*)(ws + off); off += alignup((size_t)BATCH * 2 * IMGP * IMGP * 2);
    _Float16* actA = (_Float16*)(ws + off); off += alignup((size_t)BATCH * 64 * IMGP * IMGP * 2);
    _Float16* actB = (_Float16*)(ws + off); off += alignup((size_t)BATCH * 64 * IMGP * IMGP * 2);
    _Float16* w16  = (_Float16*)(ws + off); off += alignup((size_t)627840 * 2);
    float* zpadf   = (float*)(ws + off);    off += alignup(64 * sizeof(float));
    float* h1      = (float*)(ws + off);    off += alignup((size_t)BATCH * 1024 * 4);
    float* cnnH    = (float*)(ws + off);    off += alignup((size_t)BATCH * 9 * 4);
    float* hyps    = (float*)(ws + off);    off += alignup((size_t)BATCH * 256 * 9 * 4);
    float* scores  = (float*)(ws + off);    off += alignup((size_t)BATCH * 256 * 4);
    const _Float16* zpad = (const _Float16*)zpadf;

    // ---- zero the OOB pad region (deterministic each call)
    zero_region<<<1, 64, 0, stream>>>(zpadf);

    // ---- input to NHWC f16
    {
        int n = BATCH * 2 * IMGP * IMGP;
        cvt_x_nhwc<<<(n + 255) / 256, 256, 0, stream>>>(x, x16, n);
    }

    // ---- conv weights to f16 [O][r][Cin]
    static const int Cin_[8]  = { 2, 64, 64, 64, 64, 128, 128, 128 };
    static const int Cout_[8] = { 64, 64, 64, 64, 128, 128, 128, 128 };
    static const int Hin_[8]  = { 128, 128, 64, 64, 32, 32, 16, 16 };
    size_t woff[9]; woff[0] = 0;
    for (int li = 0; li < 8; ++li) woff[li + 1] = woff[li] + (size_t)Cout_[li] * Cin_[li] * 9;
    for (int li = 0; li < 8; ++li) {
        int n = Cout_[li] * Cin_[li] * 9;
        repack_w<<<(n + 255) / 256, 256, 0, stream>>>(
            (const float*)d_in[3 + 4 * li], w16 + woff[li], Cin_[li], n);
    }

    // ---- conv stack, NHWC ping-pong:
    //   L0: x16->A   L1: A->B, pool B->A(64)
    //   L2: A->B     L3: B->A, pool A->B(32)
    //   L4: B->A     L5: A->B, pool B->A(16)
    //   L6: A->B     L7: B->A  => features (NHWC) in actA
    const _Float16* cin[8]  = { x16, actA, actA, actB, actB, actA, actA, actB };
    _Float16*       cout[8] = { actA, actB, actB, actA, actA, actB, actB, actA };
    for (int li = 0; li < 8; ++li) {
        int H = Hin_[li];
        int npix = BATCH * H * H;
        dim3 grid(npix / 64, Cout_[li] / 64);
        const float* pcb = (const float*)d_in[4 + 4 * li];
        const float* pbg = (const float*)d_in[5 + 4 * li];
        const float* pbb = (const float*)d_in[6 + 4 * li];
        if (li == 0) {
            conv3x3_wmma_generic<<<grid, 128, 0, stream>>>(
                cin[li], cout[li], w16 + woff[li], pcb, pbg, pbb,
                Cin_[li], Cout_[li], H, H);
        } else {
            int cishift = (Cin_[li] == 64) ? 1 : 2;   // steps per tap = Cin/32
            conv3x3_wmma_nhwc<<<grid, 128, 0, stream>>>(
                cin[li], cout[li], w16 + woff[li], zpad, pcb, pbg, pbb,
                Cin_[li], Cout_[li], H, H, cishift);
        }
        if (li == 1) {
            size_t n = (size_t)BATCH * 64 * 64 * 64;
            maxpool2_nhwc<<<(int)((n + 255) / 256), 256, 0, stream>>>(actB, actA, 64, 128);
        } else if (li == 3) {
            size_t n = (size_t)BATCH * 64 * 32 * 32;
            maxpool2_nhwc<<<(int)((n + 255) / 256), 256, 0, stream>>>(actA, actB, 64, 64);
        } else if (li == 5) {
            size_t n = (size_t)BATCH * 128 * 16 * 16;
            maxpool2_nhwc<<<(int)((n + 255) / 256), 256, 0, stream>>>(actB, actA, 128, 32);
        }
    }

    // ---- repack features NHWC -> NCHW flat (reuse x16 buffer; x16 is dead now)
    {
        int n = BATCH * 128 * 16 * 16;
        repack_feat<<<(n + 255) / 256, 256, 0, stream>>>(actA, x16, n);
    }

    // ---- FC head
    fc1_wmma<<<64, 128, 0, stream>>>(x16, fc1_w, fc1_b, h1);
    fc2_head<<<1, 128, 0, stream>>>(h1, fc2_w, fc2_b, cnnH);

    // ---- DSAC branch
    dlt_jacobi<<<16, 256, 0, stream>>>(src_pts, dst_pts, hyps);
    dsac_score<<<BATCH * 32, 256, 0, stream>>>(x, hyps, scores);
    dsac_select<<<1, 32, 0, stream>>>(scores, hyps, cnnH, out);
}
